// GNN_37082747634058
// MI455X (gfx1250) — compile-verified, matrix operand order checked by reading, not run on hardware
//
#include <hip/hip_runtime.h>
#include <hip/hip_bf16.h>

// ---------------------------------------------------------------------------
// GraphSAGE 3-layer pipeline for MI455X (gfx1250, wave32, WMMA).
//   per call: pack weights -> per layer { zero agg -> atomic scatter-add ->
//             fused WMMA GEMM: out = act( mean @ Wl + bl + x_dst @ Wr ) }
// ---------------------------------------------------------------------------

typedef __attribute__((ext_vector_type(16))) __bf16        v16bf;
typedef __attribute__((ext_vector_type(8)))  float         v8f;
typedef __attribute__((ext_vector_type(8)))  unsigned int  v8u;

// ----------------------------- problem sizes -------------------------------
#define N0 1363968
#define N1 123904
#define N2 11264
#define N3 1024
#define E0 1239040
#define E1 112640
#define E2 10240
#define D_IN 128
#define D_H  256
#define D_OUT 64

// workspace layout (float offsets). Region A ([0, AGG0_END)) is reused by
// layer-1/2 scratch once the layer-0 GEMM has consumed agg0/cnt0.
#define OFF_AGG0 0ULL
#define OFF_CNT0 (OFF_AGG0 + (unsigned long long)N1 * D_IN)        // 15,859,712
#define AGG0_END (OFF_CNT0 + N1)                                   // 15,983,616
#define OFF_H0   AGG0_END                                          // N1 x 256
#define WS_END   (OFF_H0 + (unsigned long long)N1 * D_H)           // 47,703,040 floats

#define OFF_AGG1 0ULL                                              // aliases agg0
#define OFF_CNT1 (OFF_AGG1 + (unsigned long long)N2 * D_H)         // 2,883,584
#define AGG1_END (OFF_CNT1 + N2)                                   // 2,894,848
#define OFF_H1   AGG1_END                                          // N2 x 256
#define OFF_AGG2 (OFF_H1 + (unsigned long long)N2 * D_H)           // 5,778,432
#define OFF_CNT2 (OFF_AGG2 + (unsigned long long)N3 * D_H)         // 6,040,576
#define AGG2_END (OFF_CNT2 + N3)                                   // 6,041,600 (< AGG0_END)

// packed bf16 weight fragments appended after WS_END (all 32B-aligned;
// sizes counted in floats = bf16 elems / 2)
#define OFF_PL0 (WS_END +      0ULL)    // 128*256 bf16 -> 16384 floats
#define OFF_PR0 (WS_END +  16384ULL)
#define OFF_PL1 (WS_END +  32768ULL)    // 256*256 bf16 -> 32768 floats
#define OFF_PR1 (WS_END +  65536ULL)
#define OFF_PL2 (WS_END +  98304ULL)    // 256*64  bf16 ->  8192 floats
#define OFF_PR2 (WS_END + 106496ULL)

// ------------------------------- kernels -----------------------------------

__global__ void zero_f32(float* __restrict__ p, long long n) {
    long long i = (long long)blockIdx.x * blockDim.x + threadIdx.x;
    if (i < n) p[i] = 0.0f;
}

// One thread per (edge, 4-float feature chunk): gather src row chunk,
// atomic-add into agg[dst], count edge once. Pure HBM-bandwidth work.
__global__ void scatter_add(const float* __restrict__ x,
                            const int* __restrict__ src,
                            const int* __restrict__ dst,
                            float* __restrict__ agg,
                            float* __restrict__ cnt,
                            int nE, int D) {
    const int vec = D >> 2;
    long long tid = (long long)blockIdx.x * blockDim.x + threadIdx.x;
    long long total = (long long)nE * vec;
    if (tid >= total) return;
    int e = (int)(tid / vec);
    int c = (int)(tid % vec);
    int s = src[e];
    int d = dst[e];
    const float4 v = reinterpret_cast<const float4*>(x + (size_t)s * D)[c];
    float* p = agg + (size_t)d * D + (size_t)c * 4;
    atomicAdd(p + 0, v.x);
    atomicAdd(p + 1, v.y);
    atomicAdd(p + 2, v.z);
    atomicAdd(p + 3, v.w);
    if (c == 0) atomicAdd(cnt + d, 1.0f);
}

// Pack fp32 row-major W[K x N] into bf16 WMMA B-fragments.
// Fragment layout (ISA 7.12.2, 16-bit B 32x16): per n-tile t, k-block kb,
// lane holds column n = t*16 + lane%16; elems e=0..7 -> K = kb*32 + cb + e,
// e=8..15 -> K = kb*32 + 16 + cb + (e-8), cb = (lane>=16)*8.
// Linear index: P[ ((t*(K/32) + kb)*32 + lane)*16 + e ]  (contiguous 32B/lane)
__global__ void pack_w_bf16(const float* __restrict__ W,
                            __bf16* __restrict__ P, int K, int N) {
    long long i = (long long)blockIdx.x * blockDim.x + threadIdx.x;
    long long total = (long long)K * N;
    if (i >= total) return;
    int e    = (int)(i & 15);
    int lane = (int)((i >> 4) & 31);
    long long blk = i >> 9;                // t*(K/32) + kb
    int kblocks = K >> 5;
    int t  = (int)(blk / kblocks);
    int kb = (int)(blk - (long long)t * kblocks);
    int cb = (lane >> 4) * 8;
    int kk = (e < 8) ? (cb + e) : (16 + cb + (e - 8));
    int k  = kb * 32 + kk;
    int n  = t * 16 + (lane & 15);
    P[i] = (__bf16)W[(size_t)k * N + n];
}

__device__ __forceinline__ __bf16 to_bf16(float f) { return (__bf16)f; }

// A-matrix (16x32 bf16) lane layout, ISA 7.12.2: lane holds row M=lane%16;
// elems e=0..7 -> K = kb + cb + e ; e=8..15 -> K = kb + 16 + cb + (e-8).
__device__ __forceinline__ v16bf load_a_pack(const float* __restrict__ row,
                                             int kb, int cb, float scale) {
    v16bf a;
#pragma unroll
    for (int e = 0; e < 8; ++e) a[e] = to_bf16(row[kb + cb + e] * scale);
#pragma unroll
    for (int e = 0; e < 8; ++e) a[8 + e] = to_bf16(row[kb + 16 + cb + e] * scale);
    return a;
}

__device__ __forceinline__ v16bf load_b_frag(const __bf16* __restrict__ P,
                                             int t, int kb, int kblocks, int lane) {
    const __bf16* bp = P + ((((size_t)t * kblocks + kb) * 32 + lane) << 4);
    return __builtin_bit_cast(v16bf, *reinterpret_cast<const v8u*>(bp));
}

// Fused SAGE layer GEMM: out[M,N] = act( (agg/cnt) @ Wl + bl + xdst @ Wr ).
// One wave32 per 16x64 output strip (4 accumulators, A fragment reused 4x).
// blockDim=128 (4 waves). N multiple of 64, M multiple of 16, K multiple of
// 32. Row guard is wave-uniform -> EXEC all-1s at every v_wmma.
__global__ void sage_gemm(const float* __restrict__ agg,
                          const float* __restrict__ cnt,
                          const float* __restrict__ xdst,
                          const __bf16* __restrict__ Pl,
                          const float* __restrict__ bl,
                          const __bf16* __restrict__ Pr,
                          float* __restrict__ out,
                          int M, int N, int K, int relu) {
    const int lane = threadIdx.x & 31;
    const int wave = threadIdx.x >> 5;
    const int wpr  = N >> 6;                 // waves per 16-row strip
    const int w    = blockIdx.x * 4 + wave;
    const int rows = M >> 4;
    const int row  = w / wpr;
    const int nb   = w - row * wpr;          // which 64-col block
    if (row >= rows) return;

    const int half = lane >> 4;
    const int lid  = lane & 15;
    const int cb   = half * 8;
    const int m    = row * 16 + lid;
    const int kblocks = K >> 5;
    const int t0   = nb * 4;

    v8f c[4];
#pragma unroll
    for (int t = 0; t < 4; ++t)
        c[t] = (v8f){0.f, 0.f, 0.f, 0.f, 0.f, 0.f, 0.f, 0.f};

    const float inv = 1.0f / fmaxf(cnt[m], 1.0f);
    const float* arow = agg  + (size_t)m * K;
    const float* xrow = xdst + (size_t)m * K;

    // phase 1: mean @ Wl  (1/cnt fused into the bf16 convert)
    for (int kb = 0; kb < kblocks; ++kb) {
        v16bf a = load_a_pack(arow, kb * 32, cb, inv);
#pragma unroll
        for (int t = 0; t < 4; ++t) {
            v16bf b = load_b_frag(Pl, t0 + t, kb, kblocks, lane);
            c[t] = __builtin_amdgcn_wmma_f32_16x16x32_bf16(
                       false, a, false, b, (short)0, c[t], false, false);
        }
    }
    // phase 2: x_dst @ Wr
    for (int kb = 0; kb < kblocks; ++kb) {
        v16bf a = load_a_pack(xrow, kb * 32, cb, 1.0f);
#pragma unroll
        for (int t = 0; t < 4; ++t) {
            v16bf b = load_b_frag(Pr, t0 + t, kb, kblocks, lane);
            c[t] = __builtin_amdgcn_wmma_f32_16x16x32_bf16(
                       false, a, false, b, (short)0, c[t], false, false);
        }
    }

    // epilogue: C/D layout -> lane<16: (M=r, N=lane); lane>=16: (M=8+r, N=lane-16)
#pragma unroll
    for (int t = 0; t < 4; ++t) {
        const int n = (t0 + t) * 16 + lid;
        const float bias = bl[n];
#pragma unroll
        for (int r = 0; r < 8; ++r) {
            int mr = row * 16 + half * 8 + r;
            float v = c[t][r] + bias;
            if (relu) v = fmaxf(v, 0.0f);
            out[(size_t)mr * N + n] = v;
        }
    }
}

// ------------------------------- launcher ----------------------------------

static inline int cdiv_ll(long long a, int b) { return (int)((a + b - 1) / b); }

extern "C" void kernel_launch(void* const* d_in, const int* in_sizes, int n_in,
                              void* d_out, int out_size, void* d_ws, size_t ws_size,
                              hipStream_t stream) {
    const float* x   = (const float*)d_in[0];
    const float* Wl0 = (const float*)d_in[1];
    const float* bl0 = (const float*)d_in[2];
    const float* Wr0 = (const float*)d_in[3];
    const float* Wl1 = (const float*)d_in[4];
    const float* bl1 = (const float*)d_in[5];
    const float* Wr1 = (const float*)d_in[6];
    const float* Wl2 = (const float*)d_in[7];
    const float* bl2 = (const float*)d_in[8];
    const float* Wr2 = (const float*)d_in[9];
    const int* src0 = (const int*)d_in[10];
    const int* dst0 = (const int*)d_in[11];
    const int* src1 = (const int*)d_in[12];
    const int* dst1 = (const int*)d_in[13];
    const int* src2 = (const int*)d_in[14];
    const int* dst2 = (const int*)d_in[15];

    float* ws = (float*)d_ws;
    float* agg0 = ws + OFF_AGG0;
    float* cnt0 = ws + OFF_CNT0;
    float* h0   = ws + OFF_H0;
    float* agg1 = ws + OFF_AGG1;
    float* cnt1 = ws + OFF_CNT1;
    float* h1   = ws + OFF_H1;
    float* agg2 = ws + OFF_AGG2;
    float* cnt2 = ws + OFF_CNT2;
    __bf16* Pl0 = (__bf16*)(ws + OFF_PL0);
    __bf16* Pr0 = (__bf16*)(ws + OFF_PR0);
    __bf16* Pl1 = (__bf16*)(ws + OFF_PL1);
    __bf16* Pr1 = (__bf16*)(ws + OFF_PR1);
    __bf16* Pl2 = (__bf16*)(ws + OFF_PL2);
    __bf16* Pr2 = (__bf16*)(ws + OFF_PR2);
    float* out  = (float*)d_out;

    const int ZB = 256, SB = 256, GB = 128, PB = 256;

    // ---- pack all weights into bf16 WMMA B-fragments (tiny) ----
    pack_w_bf16<<<cdiv_ll((long long)D_IN * D_H, PB), PB, 0, stream>>>(Wl0, Pl0, D_IN, D_H);
    pack_w_bf16<<<cdiv_ll((long long)D_IN * D_H, PB), PB, 0, stream>>>(Wr0, Pr0, D_IN, D_H);
    pack_w_bf16<<<cdiv_ll((long long)D_H  * D_H, PB), PB, 0, stream>>>(Wl1, Pl1, D_H,  D_H);
    pack_w_bf16<<<cdiv_ll((long long)D_H  * D_H, PB), PB, 0, stream>>>(Wr1, Pr1, D_H,  D_H);
    pack_w_bf16<<<cdiv_ll((long long)D_H  * D_OUT, PB), PB, 0, stream>>>(Wl2, Pl2, D_H, D_OUT);
    pack_w_bf16<<<cdiv_ll((long long)D_H  * D_OUT, PB), PB, 0, stream>>>(Wr2, Pr2, D_H, D_OUT);

    // ---- layer 0: x -> h0 (N1 x 256), relu ----
    {
        long long nz = (long long)AGG0_END;   // agg0 + cnt0 contiguous
        zero_f32<<<cdiv_ll(nz, ZB), ZB, 0, stream>>>(agg0, nz);
        long long nt = (long long)E0 * (D_IN >> 2);
        scatter_add<<<cdiv_ll(nt, SB), SB, 0, stream>>>(x, src0, dst0, agg0, cnt0, E0, D_IN);
        int waves = (N1 >> 4) * (D_H >> 6);
        sage_gemm<<<(waves + 3) / 4, GB, 0, stream>>>(
            agg0, cnt0, x, Pl0, bl0, Pr0, h0, N1, D_H, D_IN, /*relu=*/1);
    }
    // ---- layer 1: h0 -> h1 (N2 x 256), no relu ----
    {
        long long nz = (long long)AGG1_END;   // agg1 + cnt1 contiguous
        zero_f32<<<cdiv_ll(nz, ZB), ZB, 0, stream>>>(agg1, nz);
        long long nt = (long long)E1 * (D_H >> 2);
        scatter_add<<<cdiv_ll(nt, SB), SB, 0, stream>>>(h0, src1, dst1, agg1, cnt1, E1, D_H);
        int waves = (N2 >> 4) * (D_H >> 6);
        sage_gemm<<<(waves + 3) / 4, GB, 0, stream>>>(
            agg1, cnt1, h0, Pl1, bl1, Pr1, h1, N2, D_H, D_H, /*relu=*/0);
    }
    // ---- layer 2: h1 -> out (N3 x 64), relu ----
    {
        long long nz = (long long)(AGG2_END - OFF_AGG2);  // agg2 + cnt2 contiguous
        zero_f32<<<cdiv_ll(nz, ZB), ZB, 0, stream>>>(agg2, nz);
        long long nt = (long long)E2 * (D_H >> 2);
        scatter_add<<<cdiv_ll(nt, SB), SB, 0, stream>>>(h1, src2, dst2, agg2, cnt2, E2, D_H);
        int waves = (N3 >> 4) * (D_OUT >> 6);
        sage_gemm<<<(waves + 3) / 4, GB, 0, stream>>>(
            agg2, cnt2, h1, Pl2, bl2, Pr2, out, N3, D_OUT, D_H, /*relu=*/1);
    }
    (void)in_sizes; (void)n_in; (void)out_size; (void)ws_size;
}
